// ContinuousGaussian_71897752535380
// MI455X (gfx1250) — matrix operand order repaired and checked by reading, not compiled, not run on hardware
//
#include <hip/hip_runtime.h>
#include <hip/hip_bf16.h>
#include <cstdint>
#include <cstddef>

typedef _Float16 h16;
typedef __attribute__((ext_vector_type(16))) _Float16 v16h;
typedef __attribute__((ext_vector_type(8)))  _Float16 v8h;
typedef __attribute__((ext_vector_type(8)))  float    v8f;

#define BM 128
#define BN 64
#define BK 32
#define LDSS 40   // padded LDS row stride in halfs (80B: 16B-aligned, conflict-free)

#define NPIX 4096     // 64x64 feature pixels
#define HOUT 96
#define WOUT 96
#define NGAU 730
#define NGAU_PAD 768

// ---------------------------------------------------------------------------
// GAU_DICT generation (module constant in the reference, not a harness input)
// ---------------------------------------------------------------------------
__device__ const float CG_CHO1[9] = {0.0f,0.41f,0.62f,0.98f,1.13f,1.29f,1.64f,1.85f,2.36f};
__device__ const float CG_CHO2[9] = {-0.86f,-0.36f,-0.16f,0.19f,0.34f,0.49f,0.84f,1.04f,1.54f};
__device__ const float CG_CHO3[9] = {0.0f,0.33f,0.53f,0.88f,1.03f,1.18f,1.53f,1.73f,2.23f};

__global__ void cg_gdict_kernel(h16* __restrict__ gH /*[768][32]*/, float* __restrict__ gF /*[768][4]*/) {
    int idx = blockIdx.x * 256 + threadIdx.x;
    if (idx >= NGAU_PAD) return;
    float g0 = 0.f, g1 = 0.f, g2 = 0.f;
    if (idx < 729) {
        int i = idx / 81, j = (idx / 9) % 9, k = idx % 9;
        g0 = CG_CHO1[i]; g1 = CG_CHO2[j]; g2 = CG_CHO3[k];
    }
    h16* r = gH + idx * 32;
    r[0] = (h16)g0; r[1] = (h16)g1; r[2] = (h16)g2;
    #pragma unroll
    for (int t = 3; t < 32; ++t) r[t] = (h16)0.f;
    gF[idx*4+0] = g0; gF[idx*4+1] = g1; gF[idx*4+2] = g2; gF[idx*4+3] = 0.f;
}

// ---------------------------------------------------------------------------
// Weight packing: f32 [K,N] row-major -> f16 [Kpad,Npad] zero-padded
// ---------------------------------------------------------------------------
__global__ void cg_packB_kernel(const float* __restrict__ W, h16* __restrict__ out,
                                int K, int N, int Kpad, int Npad) {
    int idx = blockIdx.x * 256 + threadIdx.x;
    if (idx >= Kpad * Npad) return;
    int k = idx / Npad, n = idx % Npad;
    out[idx] = (k < K && n < N) ? (h16)W[(size_t)k * N + n] : (h16)0.f;
}

// conv1_w (O=512, I=256, 3, 3) -> B[(i*9+kh*3+kw)][o], K=2304, N=512
__global__ void cg_packConv1_kernel(const float* __restrict__ W, h16* __restrict__ out) {
    int idx = blockIdx.x * 256 + threadIdx.x;          // 2304*512
    if (idx >= 2304 * 512) return;
    int k = idx / 512, o = idx % 512;
    int i = k / 9, r = k % 9;
    out[idx] = (h16)W[(size_t)o * 2304 + i * 9 + r];
}

// ---------------------------------------------------------------------------
// Fused enc conv (3x3, 3->1024, SAME, 32x32) + pixel_shuffle(4) + unshuffle(2)
// writes fc f16 [4096,256] and leaky(feat) f16 [256][4096]
// ---------------------------------------------------------------------------
__global__ void cg_conv_enc_kernel(const float* __restrict__ inp, const float* __restrict__ w,
                                   const float* __restrict__ b,
                                   h16* __restrict__ fcH, h16* __restrict__ featLkH) {
    int idx = blockIdx.x * 256 + threadIdx.x;          // 256*64*64
    if (idx >= 256 * NPIX) return;
    int ch = idx >> 12;
    int n  = idx & 4095;
    int Hf = n >> 6, Wf = n & 63;
    int c = ch >> 2, pi = (ch >> 1) & 1, pj = ch & 1;
    int hp = 2 * Hf + pi, wp = 2 * Wf + pj;            // 0..127
    int oc = c * 16 + (hp & 3) * 4 + (wp & 3);         // conv out channel
    int hy = hp >> 2, wy = wp >> 2;                    // 0..31
    float acc = b[oc];
    const float* wk = w + (size_t)oc * 27;
    #pragma unroll
    for (int ci = 0; ci < 3; ++ci)
        #pragma unroll
        for (int kh = 0; kh < 3; ++kh) {
            int yy = hy + kh - 1;
            if (yy < 0 || yy >= 32) continue;
            #pragma unroll
            for (int kw = 0; kw < 3; ++kw) {
                int xx = wy + kw - 1;
                if (xx < 0 || xx >= 32) continue;
                acc += inp[ci * 1024 + yy * 32 + xx] * wk[ci * 9 + kh * 3 + kw];
            }
        }
    fcH[(size_t)n * 256 + ch] = (h16)acc;
    float lk = acc < 0.f ? 0.01f * acc : acc;
    featLkH[(size_t)ch * NPIX + n] = (h16)lk;
}

// ---------------------------------------------------------------------------
// im2col for conv1: [4096, 2304] f16 from featLk [256][64][64]
// ---------------------------------------------------------------------------
__global__ void cg_im2col_kernel(const h16* __restrict__ featLkH, h16* __restrict__ Acol) {
    int idx = blockIdx.x * 256 + threadIdx.x;          // 4096*2304
    if (idx >= NPIX * 2304) return;
    int n = idx / 2304, k = idx % 2304;
    int i = k / 9, r = k % 9, kh = r / 3, kw = r % 3;
    int H = n >> 6, W = n & 63;
    int yy = H + kh - 1, xx = W + kw - 1;
    h16 v = (h16)0.f;
    if (yy >= 0 && yy < 64 && xx >= 0 && xx < 64)
        v = featLkH[(size_t)i * NPIX + yy * 64 + xx];
    Acol[idx] = v;
}

// f16 transpose [768,512] -> [512,768]
__global__ void cg_transpose_kernel(const h16* __restrict__ in, h16* __restrict__ out) {
    int idx = blockIdx.x * 256 + threadIdx.x;          // 512*768
    if (idx >= 512 * NGAU_PAD) return;
    int r = idx / NGAU_PAD, c = idx % NGAU_PAD;
    out[idx] = in[(size_t)c * 512 + r];
}

// ---------------------------------------------------------------------------
// Generic WMMA GEMM: C[M,Npad] = act(A[M,Kpad] @ B[Kpad,Npad] + bias)
// Block tile 128x64, 8 wave32 waves (4x2), each wave computes a 32x32 tile:
// 2 A frags x 2 B frags -> 4 v_wmma_f32_16x16x32_f16 per K-step.
// Double-buffered LDS: one barrier per K-step, global loads for tile k+1
// issue into registers while tile k computes; tile k+2 is prefetched.
// ---------------------------------------------------------------------------
__global__ __launch_bounds__(256) void cg_gemm_wmma_kernel(
    const h16* __restrict__ A, const h16* __restrict__ B,
    const float* __restrict__ bias,
    h16* __restrict__ outH, float* __restrict__ outF,
    int M, int Kpad, int Npad, int Nreal, int act)
{
    __shared__ h16 As[2][BM][LDSS];   // row-major [m][k]
    __shared__ h16 Bs[2][BN][LDSS];   // transposed [n][k]

    int tid  = threadIdx.x;
    int wid  = tid >> 5, lane = tid & 31;
    int bm   = blockIdx.y * BM;
    int bn   = blockIdx.x * BN;
    int m0   = (wid >> 1) * 32;       // wave M offset in tile (0..96)
    int n0   = (wid & 1) * 32;        // wave N offset in tile (0/32)
    int l15  = lane & 15;
    int koff = (lane < 16) ? 0 : 8;   // A fragment K base
    int klo  = (lane < 16) ? 0 : 16;  // unused (B fragments use koff, lines 199-200)

    int arow = tid >> 1, aseg = (tid & 1) * 16;  // A coop: 128 rows x 2 segs x 16 halfs
    int brow = tid >> 3, bseg = (tid & 7) * 8;   // B coop: 32 rows x 8 segs x 8 halfs

    v8f acc00 = {}; v8f acc01 = {}; v8f acc10 = {}; v8f acc11 = {};

    const int nsteps = Kpad / BK;

    // prologue: stage tile 0 into buffer 0
    {
        v8h a0 = *(const v8h*)&A[(size_t)(bm + arow) * Kpad + aseg];
        v8h a1 = *(const v8h*)&A[(size_t)(bm + arow) * Kpad + aseg + 8];
        v8h bv = *(const v8h*)&B[(size_t)brow * Npad + bn + bseg];
        *(v8h*)&As[0][arow][aseg]     = a0;
        *(v8h*)&As[0][arow][aseg + 8] = a1;
        #pragma unroll
        for (int j = 0; j < 8; ++j) Bs[0][bseg + j][brow] = bv[j];
    }

    for (int i = 0; i < nsteps; ++i) {
        int cur = i & 1;
        bool have_next = (i + 1 < nsteps);
        v8h na0, na1, nbv;
        if (have_next) {
            int k0 = (i + 1) * BK;
            na0 = *(const v8h*)&A[(size_t)(bm + arow) * Kpad + k0 + aseg];
            na1 = *(const v8h*)&A[(size_t)(bm + arow) * Kpad + k0 + aseg + 8];
            nbv = *(const v8h*)&B[(size_t)(k0 + brow) * Npad + bn + bseg];
            if (i + 2 < nsteps) {
                int k2 = (i + 2) * BK;
                __builtin_prefetch(&A[(size_t)(bm + arow) * Kpad + k2 + aseg], 0, 1);
                __builtin_prefetch(&B[(size_t)(k2 + brow) * Npad + bn + bseg], 0, 1);
            }
        }

        __syncthreads();   // tile `cur` fully staged; prior reads of `1-cur` done

        // A fragments (16x32 f16 layout): lanes 0-15 K{0-7,16-23}, lanes 16-31 K{8-15,24-31}
        v8h a0lo = *(const v8h*)&As[cur][m0 + l15][koff];
        v8h a0hi = *(const v8h*)&As[cur][m0 + l15][16 + koff];
        v8h a1lo = *(const v8h*)&As[cur][m0 + 16 + l15][koff];
        v8h a1hi = *(const v8h*)&As[cur][m0 + 16 + l15][16 + koff];
        v16h af0, af1;
        #pragma unroll
        for (int e = 0; e < 8; ++e) {
            af0[e] = a0lo[e]; af0[8 + e] = a0hi[e];
            af1[e] = a1lo[e]; af1[8 + e] = a1hi[e];
        }

        // B fragments (32x16): lane holds column n, K{koff..koff+7, 16+koff..16+koff+7} (matches A)
        const h16* bp0 = &Bs[cur][n0 + l15][koff];
        const h16* bp1 = &Bs[cur][n0 + 16 + l15][koff];
        v8h b0lo = *(const v8h*)bp0;       v8h b0hi = *(const v8h*)(bp0 + 16);
        v8h b1lo = *(const v8h*)bp1;       v8h b1hi = *(const v8h*)(bp1 + 16);
        v16h bf0, bf1;
        #pragma unroll
        for (int e = 0; e < 8; ++e) {
            bf0[e] = b0lo[e]; bf0[8 + e] = b0hi[e];
            bf1[e] = b1lo[e]; bf1[8 + e] = b1hi[e];
        }

        acc00 = __builtin_amdgcn_wmma_f32_16x16x32_f16(false, af0, false, bf0, (short)0, acc00, false, false);
        acc01 = __builtin_amdgcn_wmma_f32_16x16x32_f16(false, af0, false, bf1, (short)0, acc01, false, false);
        acc10 = __builtin_amdgcn_wmma_f32_16x16x32_f16(false, af1, false, bf0, (short)0, acc10, false, false);
        acc11 = __builtin_amdgcn_wmma_f32_16x16x32_f16(false, af1, false, bf1, (short)0, acc11, false, false);

        if (have_next) {   // stage tile i+1 into the other buffer (consumed after next barrier)
            int nxt = 1 - cur;
            *(v8h*)&As[nxt][arow][aseg]     = na0;
            *(v8h*)&As[nxt][arow][aseg + 8] = na1;
            #pragma unroll
            for (int j = 0; j < 8; ++j) Bs[nxt][bseg + j][brow] = nbv[j];
        }
    }

    // Epilogue: C layout — VGPR r: lanes 0-15 M=r, lanes 16-31 M=8+r; N=lane&15
    int rowb = bm + m0 + ((lane < 16) ? 0 : 8);
    int col0 = bn + n0 + l15;
    int col1 = col0 + 16;
    float bv0 = (bias && col0 < Nreal) ? bias[col0] : 0.f;
    float bv1 = (bias && col1 < Nreal) ? bias[col1] : 0.f;
    #pragma unroll
    for (int t = 0; t < 2; ++t) {
        v8f a0 = t ? acc10 : acc00;
        v8f a1 = t ? acc11 : acc01;
        int rb = rowb + t * 16;
        #pragma unroll
        for (int r = 0; r < 8; ++r) {
            int row = rb + r;
            float v0 = a0[r] + bv0;
            float v1 = a1[r] + bv1;
            if (act == 1)      { v0 = fmaxf(v0, 0.f); v1 = fmaxf(v1, 0.f); }
            else if (act == 2) { v0 = tanhf(v0);      v1 = tanhf(v1); }
            size_t o0 = (size_t)row * Npad + col0;
            size_t o1 = (size_t)row * Npad + col1;
            if (outH) { outH[o0] = (h16)v0; outH[o1] = (h16)v1; }
            if (outF) { outF[o0] = v0;      outF[o1] = v1; }
        }
    }
}

// ---------------------------------------------------------------------------
// Fused row softmax (over 730 valid dict entries) + para = att @ GAU_DICT
// one block of 256 threads per pixel row
// ---------------------------------------------------------------------------
__global__ __launch_bounds__(256) void cg_softpara_kernel(
    const float* __restrict__ attF /*[4096][768]*/, const float* __restrict__ gF /*[768][4]*/,
    float* __restrict__ paraF /*[4096][4]*/)
{
    int n = blockIdx.x;
    int tid = threadIdx.x;
    const float* row = attF + (size_t)n * NGAU_PAD;
    __shared__ float red[8];

    float mx = -3.0e38f;
    for (int j = tid; j < NGAU; j += 256) mx = fmaxf(mx, row[j]);
    for (int o = 16; o > 0; o >>= 1) mx = fmaxf(mx, __shfl_xor(mx, o, 32));
    if ((tid & 31) == 0) red[tid >> 5] = mx;
    __syncthreads();
    float MX = fmaxf(fmaxf(fmaxf(red[0], red[1]), fmaxf(red[2], red[3])),
                     fmaxf(fmaxf(red[4], red[5]), fmaxf(red[6], red[7])));
    __syncthreads();

    float s = 0.f, p0 = 0.f, p1 = 0.f, p2 = 0.f;
    for (int j = tid; j < NGAU; j += 256) {
        float e = __expf(row[j] - MX);
        s  += e;
        p0 += e * gF[j * 4 + 0];
        p1 += e * gF[j * 4 + 1];
        p2 += e * gF[j * 4 + 2];
    }
    float vals[4] = { s, p0, p1, p2 };
    float out[4];
    #pragma unroll
    for (int q = 0; q < 4; ++q) {
        float v = vals[q];
        for (int o = 16; o > 0; o >>= 1) v += __shfl_xor(v, o, 32);
        if ((tid & 31) == 0) red[tid >> 5] = v;
        __syncthreads();
        out[q] = red[0] + red[1] + red[2] + red[3] + red[4] + red[5] + red[6] + red[7];
        __syncthreads();
    }
    if (tid == 0) {
        float inv = 1.0f / out[0];
        paraF[n * 4 + 0] = out[1] * inv;
        paraF[n * 4 + 1] = out[2] * inv;
        paraF[n * 4 + 2] = out[3] * inv;
    }
}

// ---------------------------------------------------------------------------
// Per-gaussian parameter prep: inverse covariance, pixel-space means, colors
// g8[n] = {ca, cb, cc, px, py, r, g, b}
// ---------------------------------------------------------------------------
__global__ void cg_gprep_kernel(const float* __restrict__ paraF, const float* __restrict__ colorF,
                                const float* __restrict__ offF, const int* __restrict__ scale_p,
                                float* __restrict__ g8) {
    int n = blockIdx.x * 256 + threadIdx.x;
    if (n >= NPIX) return;
    float scale = (float)scale_p[0];
    float f = 0.25f * scale;
    float l1 = paraF[n * 4 + 0] * f;
    float l2 = paraF[n * 4 + 1] * f;
    float l3 = paraF[n * 4 + 2] * f;
    float c00 = l1 * l1, c01 = l1 * l2, c11 = l2 * l2 + l3 * l3;
    float det = c00 * c11 - c01 * c01;
    bool valid = det > 1e-10f;
    float inv = valid ? 1.0f / det : 0.f;
    float ca = c11 * inv, cb = -c01 * inv, cc = c00 * inv;

    float coord0 = 2.0f * (float)(n & 63) / 64.0f - 1.0f;   // coord[:,0] (yg)
    float coord1 = 2.0f * (float)(n >> 6) / 64.0f - 1.0f;   // coord[:,1] (xg)
    float off0 = offF[(size_t)n * 64 + 0];
    float off1 = offF[(size_t)n * 64 + 1];
    float mx = coord0 + 2.0f * off0 / 32.0f - 1.0f / (float)WOUT;
    float my = coord1 + 2.0f * off1 / 32.0f - 1.0f / (float)HOUT;
    float px = 0.5f * (float)WOUT * (mx + 1.0f) - 0.5f;
    float py = 0.5f * (float)HOUT * (my + 1.0f) - 0.5f;

    float cr = valid ? colorF[(size_t)n * 64 + 0] : 0.f;
    float cg = valid ? colorF[(size_t)n * 64 + 1] : 0.f;
    float cbl= valid ? colorF[(size_t)n * 64 + 2] : 0.f;

    float* o = g8 + (size_t)n * 8;
    o[0] = ca; o[1] = cb; o[2] = cc; o[3] = px; o[4] = py;
    o[5] = cr; o[6] = cg; o[7] = cbl;
}

// ---------------------------------------------------------------------------
// Raster: out[c][y][x] = sum_n alpha(n; x,y) * color[n][c]
// ---------------------------------------------------------------------------
__global__ __launch_bounds__(256) void cg_raster_kernel(const float* __restrict__ g8,
                                                        float* __restrict__ outp) {
    __shared__ float sh[256 * 8];
    int p = blockIdx.x * 256 + threadIdx.x;     // 9216 pixels
    float fx = (float)(p % WOUT);
    float fy = (float)(p / WOUT);
    float accR = 0.f, accG = 0.f, accB = 0.f;

    for (int base = 0; base < NPIX; base += 256) {
        __syncthreads();
        const float* src = g8 + (size_t)(base + threadIdx.x) * 8;
        #pragma unroll
        for (int j = 0; j < 8; ++j) sh[threadIdx.x * 8 + j] = src[j];
        __syncthreads();
        for (int q = 0; q < 256; ++q) {
            const float* gp = &sh[q * 8];
            float dx = fx - gp[3], dy = fy - gp[4];
            float sig = 0.5f * (gp[0] * dx * dx + gp[2] * dy * dy) + gp[1] * dx * dy;
            if (sig >= 0.f) {
                float al = __expf(-sig);
                accR += al * gp[5];
                accG += al * gp[6];
                accB += al * gp[7];
            }
        }
    }
    outp[p]                       = accR;
    outp[HOUT * WOUT + p]         = accG;
    outp[2 * HOUT * WOUT + p]     = accB;
}

// ---------------------------------------------------------------------------
// Host launcher
// ---------------------------------------------------------------------------
extern "C" void kernel_launch(void* const* d_in, const int* in_sizes, int n_in,
                              void* d_out, int out_size, void* d_ws, size_t ws_size,
                              hipStream_t stream) {
    (void)in_sizes; (void)n_in; (void)out_size; (void)ws_size;

    const float* inp   = (const float*)d_in[0];
    const int*   scale = (const int*)  d_in[1];
    const float* enc_w = (const float*)d_in[2];
    const float* enc_b = (const float*)d_in[3];
    const float* c1_w  = (const float*)d_in[4];
    const float* c1_b  = (const float*)d_in[5];
    // mlpv: 5 layers @ d_in[6..15]; mlpc: 6 layers @ [16..27]; mlpo: 6 layers @ [28..39]

    // ---- workspace bump allocator (256B aligned) ----
    char* ws = (char*)d_ws; size_t off = 0;
    auto alloc = [&](size_t bytes) -> void* {
        off = (off + 255) & ~(size_t)255;
        void* p = ws + off; off += bytes; return p;
    };

    h16*   fcH     = (h16*)  alloc((size_t)NPIX * 256 * 2);
    h16*   featLkH = (h16*)  alloc((size_t)256 * NPIX * 2);
    h16*   AcolH   = (h16*)  alloc((size_t)NPIX * 2304 * 2);
    h16*   Wc1p    = (h16*)  alloc((size_t)2304 * 512 * 2);
    h16*   pfH     = (h16*)  alloc((size_t)NPIX * 512 * 2);
    h16*   gdictH  = (h16*)  alloc((size_t)NGAU_PAD * 32 * 2);
    float* gdictF  = (float*)alloc((size_t)NGAU_PAD * 4 * 4);
    h16*   vA      = (h16*)  alloc((size_t)NGAU_PAD * 512 * 2);
    h16*   vB      = (h16*)  alloc((size_t)NGAU_PAD * 512 * 2);
    h16*   vecTH   = (h16*)  alloc((size_t)512 * NGAU_PAD * 2);
    float* attF    = (float*)alloc((size_t)NPIX * NGAU_PAD * 4);
    h16*   actA    = (h16*)  alloc((size_t)NPIX * 1024 * 2);
    h16*   actB    = (h16*)  alloc((size_t)NPIX * 1024 * 2);
    float* colorF  = (float*)alloc((size_t)NPIX * 64 * 4);
    float* offF    = (float*)alloc((size_t)NPIX * 64 * 4);
    float* paraF   = (float*)alloc((size_t)NPIX * 4 * 4);
    float* g8      = (float*)alloc((size_t)NPIX * 8 * 4);

    // packed weight buffers
    const int vK[5]   = {3, 256, 512, 512, 512};
    const int vKp[5]  = {32, 256, 512, 512, 512};
    const int vN[5]   = {256, 512, 512, 512, 512};
    const int cK[6]   = {256, 512, 1024, 256, 128, 64};
    const int cN[6]   = {512, 1024, 256, 128, 64, 3};
    const int cNp[6]  = {512, 1024, 256, 128, 64, 64};
    h16* wvP[5]; h16* wcP[6]; h16* woP[6];
    for (int l = 0; l < 5; ++l) wvP[l] = (h16*)alloc((size_t)vKp[l] * vN[l] * 2);
    for (int l = 0; l < 6; ++l) wcP[l] = (h16*)alloc((size_t)cK[l] * cNp[l] * 2);
    for (int l = 0; l < 6; ++l) woP[l] = (h16*)alloc((size_t)cK[l] * cNp[l] * 2);

    auto packB = [&](const void* w, h16* out, int K, int N, int Kpad, int Npad) {
        int total = Kpad * Npad;
        cg_packB_kernel<<<(total + 255) / 256, 256, 0, stream>>>(
            (const float*)w, out, K, N, Kpad, Npad);
    };
    auto gemm = [&](const void* A, const h16* B, const void* bias, h16* outH, float* outF,
                    int M, int Kpad, int Npad, int Nreal, int act) {
        dim3 g(Npad / BN, M / BM);
        cg_gemm_wmma_kernel<<<g, 256, 0, stream>>>(
            (const h16*)A, B, (const float*)bias, outH, outF, M, Kpad, Npad, Nreal, act);
    };

    // ---- stage 0: constants + packing ----
    cg_gdict_kernel<<<(NGAU_PAD + 255) / 256, 256, 0, stream>>>(gdictH, gdictF);
    cg_packConv1_kernel<<<(2304 * 512 + 255) / 256, 256, 0, stream>>>(c1_w, Wc1p);
    for (int l = 0; l < 5; ++l) packB(d_in[6 + 2 * l],  wvP[l], vK[l], vN[l], vKp[l], vN[l]);
    for (int l = 0; l < 6; ++l) packB(d_in[16 + 2 * l], wcP[l], cK[l], cN[l], cK[l], cNp[l]);
    for (int l = 0; l < 6; ++l) packB(d_in[28 + 2 * l], woP[l], cK[l], cN[l], cK[l], cNp[l]);

    // ---- stage 1: encoder conv (fused shuffles) + im2col ----
    cg_conv_enc_kernel<<<(256 * NPIX + 255) / 256, 256, 0, stream>>>(inp, enc_w, enc_b, fcH, featLkH);
    cg_im2col_kernel<<<(NPIX * 2304 + 255) / 256, 256, 0, stream>>>(featLkH, AcolH);

    // ---- stage 2: conv1 as GEMM -> pf ----
    gemm(AcolH, Wc1p, c1_b, pfH, nullptr, NPIX, 2304, 512, 512, 0);

    // ---- stage 3: mlpv over gaussian dictionary (M padded 730->768 = 6*128) ----
    {
        const h16* in = gdictH; h16* bufs[2] = { vA, vB }; int pp = 0;
        for (int l = 0; l < 5; ++l) {
            h16* o = bufs[pp];
            gemm(in, wvP[l], d_in[7 + 2 * l], o, nullptr,
                 NGAU_PAD, vKp[l], vN[l], vN[l], (l < 4) ? 1 : 0);
            in = o; pp ^= 1;
        }
        // final output landed in vA (L0->vA, L1->vB, L2->vA, L3->vB, L4->vA)
        cg_transpose_kernel<<<(512 * NGAU_PAD + 255) / 256, 256, 0, stream>>>(vA, vecTH);
    }

    // ---- stage 4: attention logits + fused softmax/para ----
    gemm(pfH, vecTH, nullptr, nullptr, attF, NPIX, 512, NGAU_PAD, NGAU_PAD, 0);
    cg_softpara_kernel<<<NPIX, 256, 0, stream>>>(attF, gdictF, paraF);

    // ---- stage 5: color MLP (mlpc) ----
    {
        const h16* in = fcH; h16* bufs[2] = { actA, actB }; int pp = 0;
        for (int l = 0; l < 5; ++l) {
            h16* o = bufs[pp];
            gemm(in, wcP[l], d_in[17 + 2 * l], o, nullptr, NPIX, cK[l], cNp[l], cN[l], 1);
            in = o; pp ^= 1;
        }
        gemm(in, wcP[5], d_in[17 + 10], nullptr, colorF, NPIX, cK[5], cNp[5], cN[5], 0);
    }

    // ---- stage 6: offset MLP (mlpo), tanh epilogue ----
    {
        const h16* in = fcH; h16* bufs[2] = { actA, actB }; int pp = 0;
        for (int l = 0; l < 5; ++l) {
            h16* o = bufs[pp];
            gemm(in, woP[l], d_in[29 + 2 * l], o, nullptr, NPIX, cK[l], cNp[l], cN[l], 1);
            in = o; pp ^= 1;
        }
        gemm(in, woP[5], d_in[29 + 10], nullptr, offF, NPIX, cK[5], cNp[5], cN[5], 2);
    }

    // ---- stage 7: gaussian prep + raster ----
    cg_gprep_kernel<<<(NPIX + 255) / 256, 256, 0, stream>>>(paraF, colorF, offF, scale, g8);
    cg_raster_kernel<<<(HOUT * WOUT) / 256, 256, 0, stream>>>(g8, (float*)d_out);
}
